// Captioner_89799176225634
// MI455X (gfx1250) — matrix operand order repaired and checked by reading, not compiled, not run on hardware
//
#include <hip/hip_runtime.h>
#include <hip/hip_bf16.h>
#include <math.h>

// ---------------------------------------------------------------------------
// Model dims (fixed by the reference)
// ---------------------------------------------------------------------------
#define BB     128      // batch
#define LL     64       // image locations
#define FF     2048     // feature dim
#define UU     512      // attention units
#define HH     512      // lstm hidden
#define EDIM   256
#define VOCAB  10000
#define TT     20
#define NSTEP  (TT - 1) // 19
#define START_TOK 1
#define XW     (EDIM + FF + HH)   // 2816 concat width

// ---------------------------------------------------------------------------
// WMMA vector types (gfx1250, wave32)
// ---------------------------------------------------------------------------
typedef __attribute__((ext_vector_type(16))) __bf16 v16bf;
typedef __attribute__((ext_vector_type(8)))  __bf16 v8bf;
typedef __attribute__((ext_vector_type(8)))  float  v8f;

__device__ __forceinline__ __bf16 f2bf(float f) {
    union { float f; unsigned u; } v; v.f = f;
    unsigned r = v.u + 0x7FFFu + ((v.u >> 16) & 1u);   // round-to-nearest-even
    unsigned short h = (unsigned short)(r >> 16);
    __bf16 b;
    __builtin_memcpy(&b, &h, 2);
    return b;
}

__device__ __forceinline__ float sigmoidf_(float x) {
    return 1.0f / (1.0f + __expf(-x));
}

// ---------------------------------------------------------------------------
// Transpose + convert: W (K x N, f32, row-major)  ->  Wt[n*ldt + kOff + k] bf16
// Standard 32x32 LDS tile transpose, both sides coalesced.
// ---------------------------------------------------------------------------
__global__ __launch_bounds__(256)
void transpose_to_bf16_kernel(const float* __restrict__ W, __bf16* __restrict__ Wt,
                              int K, int N, long long ldt, long long kOff)
{
    __shared__ __bf16 tile[32][33];
    const int tx = threadIdx.x;            // 0..31
    const int ty = threadIdx.y;            // 0..7
    const int k0 = blockIdx.y * 32;
    const int n0 = blockIdx.x * 32;
    #pragma unroll
    for (int r = 0; r < 4; ++r) {
        int k = k0 + ty + r * 8;
        float v = (k < K && (n0 + tx) < N) ? W[(long long)k * N + (n0 + tx)] : 0.0f;
        tile[ty + r * 8][tx] = f2bf(v);    // tile[k_local][n_local]
    }
    __syncthreads();
    #pragma unroll
    for (int r = 0; r < 4; ++r) {
        int n = n0 + ty + r * 8;
        int k = k0 + tx;
        if (n < N && k < K)
            Wt[(long long)n * ldt + kOff + k] = tile[tx][ty + r * 8];
    }
}

static inline void tconv(hipStream_t s, const float* W, __bf16* Wt,
                         int K, int N, long long ldt, long long kOff)
{
    dim3 grid((N + 31) / 32, (K + 31) / 32);
    transpose_to_bf16_kernel<<<grid, dim3(32, 8), 0, s>>>(W, Wt, K, N, ldt, kOff);
}

// ---------------------------------------------------------------------------
// Tiled bf16-WMMA GEMM.
//   C[m,n] = act( A(MxK) * B(KxN) + bias[n] ),  out[m*osm + n*osn + ooff]
//   A: f32 (a_bf16==0, converted while staging) or bf16 (a_bf16==1, pure copy)
//   B: pre-transposed bf16, Bt[n*ldbt + k]
//   act: 0=none 1=relu 2=tanh 3=tanh(relu)
// Tile 64x128, BK=64, 256 threads = 8 wave32 as 4(M) x 2(N); each wave owns
// a 16x64 strip -> 4 C fragments, 8 v_wmma per K-stage.
// REQUIRES: M % 64 == 0, K % 64 == 0 (true for every call in this model).
// ---------------------------------------------------------------------------
#define BM 64
#define BN 128
#define BK 64

__global__ __launch_bounds__(256)
void gemm_wmma_kernel(const void* __restrict__ Av, long long lda, int a_bf16,
                      const __bf16* __restrict__ Bt, long long ldbt,
                      const float* __restrict__ bias,
                      float* __restrict__ out,
                      int M, int N, int K, int act,
                      long long osm, long long osn, long long ooff)
{
    __shared__ __align__(16) __bf16 As[BM * BK];   // [m][k]  8 KB
    __shared__ __align__(16) __bf16 Bs[BN * BK];   // [n][k] 16 KB

    const int tid   = threadIdx.x;
    const int lane  = tid & 31;
    const int wave  = tid >> 5;
    const int m0    = blockIdx.y * BM;
    const int n0    = blockIdx.x * BN;
    const int msub  = (wave & 3) * 16;     // 4 waves along M
    const int nbase = (wave >> 2) * 64;    // 2 waves along N, 64 cols each
    const int alane = lane & 15;
    const int ahi   = lane >> 4;

    v8f acc[4];
    #pragma unroll
    for (int j = 0; j < 4; ++j) acc[j] = v8f{0.f,0.f,0.f,0.f,0.f,0.f,0.f,0.f};

    // cooperative loader indices
    const int arow  = tid >> 2;            // 0..63
    const int acol  = (tid & 3) * 16;      // 0,16,32,48
    const int bnrow = tid >> 1;            // 0..127
    const int bkoff = (tid & 1) * 32;      // 0,32
    const bool bRowOk = (n0 + bnrow) < N;  // only VOCAB tail blocks diverge

    for (int k0 = 0; k0 < K; k0 += BK) {
        // ---- stage A (64x64): pure 16B copies (bf16) or packed convert (f32)
        if (a_bf16) {
            const __bf16* ap = (const __bf16*)Av +
                               (long long)(m0 + arow) * lda + k0 + acol;
            *(v8bf*)&As[arow * BK + acol]     = *(const v8bf*)(ap);
            *(v8bf*)&As[arow * BK + acol + 8] = *(const v8bf*)(ap + 8);
            if (k0 + BK < K) __builtin_prefetch(ap + BK, 0, 3);
        } else {
            const float* ap = (const float*)Av +
                              (long long)(m0 + arow) * lda + k0 + acol;
            v8bf p0, p1;
            #pragma unroll
            for (int i = 0; i < 8; ++i) { p0[i] = f2bf(ap[i]); p1[i] = f2bf(ap[8 + i]); }
            *(v8bf*)&As[arow * BK + acol]     = p0;
            *(v8bf*)&As[arow * BK + acol + 8] = p1;
            if (k0 + BK < K) __builtin_prefetch(ap + BK, 0, 3);
        }
        // ---- stage B (128x64): pure 16B copies from pre-transposed bf16
        {
            __bf16* dst = &Bs[bnrow * BK + bkoff];
            if (bRowOk) {
                const __bf16* bp = Bt + (long long)(n0 + bnrow) * ldbt + k0 + bkoff;
                *(v8bf*)(dst)      = *(const v8bf*)(bp);
                *(v8bf*)(dst + 8)  = *(const v8bf*)(bp + 8);
                *(v8bf*)(dst + 16) = *(const v8bf*)(bp + 16);
                *(v8bf*)(dst + 24) = *(const v8bf*)(bp + 24);
                if (k0 + BK < K) __builtin_prefetch(bp + BK, 0, 3);
            } else {
                v8bf z = {};
                *(v8bf*)(dst)      = z;
                *(v8bf*)(dst + 8)  = z;
                *(v8bf*)(dst + 16) = z;
                *(v8bf*)(dst + 24) = z;
            }
        }
        __syncthreads();

        // ---- compute: 2 k-steps x 4 N-frags = 8 wmma
        #pragma unroll
        for (int kk = 0; kk < BK; kk += 32) {
            const __bf16* ab = &As[(msub + alane) * BK + kk + ahi * 8];
            v16bf afrag = __builtin_shufflevector(*(const v8bf*)ab,
                                                  *(const v8bf*)(ab + 16),
                            0,1,2,3,4,5,6,7,8,9,10,11,12,13,14,15);
            #pragma unroll
            for (int j = 0; j < 4; ++j) {
                const __bf16* bb = &Bs[(nbase + j * 16 + alane) * BK + kk + ahi * 8];
                v16bf bfrag = __builtin_shufflevector(*(const v8bf*)bb,
                                                      *(const v8bf*)(bb + 16),
                                0,1,2,3,4,5,6,7,8,9,10,11,12,13,14,15);
                acc[j] = __builtin_amdgcn_wmma_f32_16x16x32_bf16(
                             false, afrag, false, bfrag, (short)0, acc[j],
                             false, false);
            }
        }
        __syncthreads();
    }

    // ---- store: VGPR r -> M = msub + r + 8*lane_hi, N = lane&15
    const int cm = m0 + msub + ahi * 8;
    #pragma unroll
    for (int j = 0; j < 4; ++j) {
        const int n = n0 + nbase + j * 16 + alane;
        if (n >= N) continue;
        const float bv = bias ? bias[n] : 0.0f;
        #pragma unroll
        for (int r = 0; r < 8; ++r) {
            const int m = cm + r;                      // M % 64 == 0 -> in range
            long long idx = (long long)m * osm + (long long)n * osn + ooff;
            float v = acc[j][r] + bv;
            if      (act == 1) v = fmaxf(v, 0.0f);
            else if (act == 2) v = tanhf(v);
            else if (act == 3) v = tanhf(fmaxf(v, 0.0f));
            out[idx] = v;
        }
    }
}

static inline void gemm(hipStream_t s, const void* A, long long lda, int a_bf16,
                        const __bf16* Bt, long long ldbt, const float* bias,
                        float* out, int M, int N, int K, int act,
                        long long osm, long long osn, long long ooff)
{
    dim3 grid((N + BN - 1) / BN, (M + BM - 1) / BM);
    gemm_wmma_kernel<<<grid, 256, 0, s>>>(A, lda, a_bf16, Bt, ldbt, bias, out,
                                          M, N, K, act, osm, osn, ooff);
}

// ---------------------------------------------------------------------------
// mean over L:  mean[b,f] = (1/L) * sum_l img[b,l,f]
// ---------------------------------------------------------------------------
__global__ void mean_pool_kernel(const float* __restrict__ img,
                                 float* __restrict__ mean)
{
    int idx = blockIdx.x * blockDim.x + threadIdx.x;     // B*F threads
    if (idx >= BB * FF) return;
    int b = idx / FF, f = idx % FF;
    const float* p = img + ((long long)b * LL) * FF + f;
    float s = 0.f;
    #pragma unroll 8
    for (int l = 0; l < LL; ++l) s += p[(long long)l * FF];
    mean[idx] = s * (1.0f / (float)LL);
}

// ---------------------------------------------------------------------------
// seed hbf / xcat h-slot from init f32 h
// ---------------------------------------------------------------------------
__global__ void seed_h_kernel(const float* __restrict__ h,
                              __bf16* __restrict__ hbf,
                              __bf16* __restrict__ xcat)
{
    int idx = blockIdx.x * blockDim.x + threadIdx.x;     // B*H
    if (idx >= BB * HH) return;
    int b = idx / HH, u = idx % HH;
    __bf16 v = f2bf(h[idx]);
    hbf[idx] = v;
    xcat[(long long)b * XW + EDIM + FF + u] = v;
}

// ---------------------------------------------------------------------------
// embedding gather into concat buffer (bf16): xcat[b, 0..EDIM) = emb[word_b]
// ---------------------------------------------------------------------------
__global__ void embed_kernel(const int* __restrict__ target,
                             const float* __restrict__ emb,
                             __bf16* __restrict__ xcat, int step)
{
    int b = blockIdx.x;
    int j = threadIdx.x;                 // EDIM == 256 == blockDim
    int word = (step == 0) ? START_TOK : target[b * TT + step];
    xcat[(long long)b * XW + j] = f2bf(emb[(long long)word * EDIM + j]);
}

// ---------------------------------------------------------------------------
// Fused Bahdanau attention: one block per batch row (256 threads).
// Writes ctx (bf16) into xcat[b, EDIM .. EDIM+F).
// ---------------------------------------------------------------------------
__global__ __launch_bounds__(256)
void attention_kernel(const float* __restrict__ feat_proj,
                      const float* __restrict__ hid_proj,
                      const float* __restrict__ V_k,
                      const float* __restrict__ V_b,
                      const float* __restrict__ fbeta_k,
                      const float* __restrict__ fbeta_b,
                      const float* __restrict__ h,
                      const float* __restrict__ img,
                      __bf16* __restrict__ xcat)
{
    __shared__ float hp[UU];
    __shared__ float part[256];
    __shared__ float sc[LL];
    __shared__ float red[2];
    __shared__ float bsh;

    const int b = blockIdx.x;
    const int t = threadIdx.x;

    hp[t]       = hid_proj[(long long)b * UU + t];
    hp[t + 256] = hid_proj[(long long)b * UU + t + 256];
    __syncthreads();

    // score partials: 4 threads per location l
    {
        const int l   = t >> 2;
        const int seg = t & 3;
        const float* fp = feat_proj + ((long long)b * LL + l) * UU + seg * 128;
        const float* vk = V_k + seg * 128;
        const float* hh = hp + seg * 128;
        float s = 0.f;
        #pragma unroll 4
        for (int u = 0; u < 128; ++u)
            s += tanhf(fp[u] + hh[u]) * vk[u];
        part[t] = s;
    }
    __syncthreads();
    if (t < LL)
        sc[t] = part[4 * t] + part[4 * t + 1] + part[4 * t + 2] + part[4 * t + 3]
              + V_b[0];
    __syncthreads();

    // softmax over L=64
    if (t == 0) {
        float mx = sc[0];
        for (int l = 1; l < LL; ++l) mx = fmaxf(mx, sc[l]);
        red[0] = mx;
    }
    __syncthreads();
    if (t < LL) sc[t] = __expf(sc[t] - red[0]);
    __syncthreads();
    if (t == 0) {
        float sm = 0.f;
        for (int l = 0; l < LL; ++l) sm += sc[l];
        red[1] = 1.0f / sm;
    }
    __syncthreads();
    if (t < LL) sc[t] *= red[1];

    // beta = sigmoid(h . fbeta + fb)
    {
        float s = h[(long long)b * HH + 2 * t]     * fbeta_k[2 * t]
                + h[(long long)b * HH + 2 * t + 1] * fbeta_k[2 * t + 1];
        part[t] = s;
    }
    __syncthreads();
    for (int s = 128; s > 0; s >>= 1) {
        if (t < s) part[t] += part[t + s];
        __syncthreads();
    }
    if (t == 0) bsh = sigmoidf_(part[0] + fbeta_b[0]);
    __syncthreads();

    // context: ctx[f] = beta * sum_l attn[l] * img[b,l,f]
    const float beta = bsh;
    for (int f = t; f < FF; f += 256) {
        const float* ip = img + ((long long)b * LL) * FF + f;
        float acc = 0.f;
        #pragma unroll 8
        for (int l = 0; l < LL; ++l)
            acc += sc[l] * ip[(long long)l * FF];
        xcat[(long long)b * XW + EDIM + f] = f2bf(beta * acc);
    }
}

// ---------------------------------------------------------------------------
// LSTM gate nonlinearity (Keras order i,f,g,o); updates h(f32), c(f32),
// hbf(bf16) and xcat h-slot (bf16).
// ---------------------------------------------------------------------------
__global__ void lstm_gate_kernel(const float* __restrict__ z,
                                 float* __restrict__ h,
                                 float* __restrict__ c,
                                 __bf16* __restrict__ hbf,
                                 __bf16* __restrict__ xcat)
{
    int idx = blockIdx.x * blockDim.x + threadIdx.x;    // B*H
    if (idx >= BB * HH) return;
    int b = idx / HH, u = idx % HH;
    const float* zb = z + (long long)b * (4 * HH);
    float zi = zb[u];
    float zf = zb[HH + u];
    float zg = zb[2 * HH + u];
    float zo = zb[3 * HH + u];
    float c2 = sigmoidf_(zf) * c[idx] + sigmoidf_(zi) * tanhf(zg);
    float h2 = sigmoidf_(zo) * tanhf(c2);
    c[idx] = c2;
    h[idx] = h2;
    __bf16 hb = f2bf(h2);
    hbf[idx] = hb;
    xcat[(long long)b * XW + EDIM + FF + u] = hb;
}

// ---------------------------------------------------------------------------
// Host orchestration
// ---------------------------------------------------------------------------
extern "C" void kernel_launch(void* const* d_in, const int* in_sizes, int n_in,
                              void* d_out, int out_size, void* d_ws, size_t ws_size,
                              hipStream_t stream)
{
    (void)in_sizes; (void)n_in; (void)out_size; (void)ws_size;

    const float* img      = (const float*)d_in[0];
    const int*   target   = (const int*)  d_in[1];
    const float* emb      = (const float*)d_in[2];
    const float* W1_k     = (const float*)d_in[3];
    const float* W1_b     = (const float*)d_in[4];
    const float* W2_k     = (const float*)d_in[5];
    const float* W2_b     = (const float*)d_in[6];
    const float* V_k      = (const float*)d_in[7];
    const float* V_b      = (const float*)d_in[8];
    const float* fbeta_k  = (const float*)d_in[9];
    const float* fbeta_b  = (const float*)d_in[10];
    const float* lstm_k   = (const float*)d_in[11];
    const float* lstm_rk  = (const float*)d_in[12];
    const float* lstm_b   = (const float*)d_in[13];
    const float* logit_k  = (const float*)d_in[14];
    const float* logit_b  = (const float*)d_in[15];
    const float* ih0_k    = (const float*)d_in[16];
    const float* ih0_b    = (const float*)d_in[17];
    const float* ih1_k    = (const float*)d_in[18];
    const float* ih1_b    = (const float*)d_in[19];
    const float* ic0_k    = (const float*)d_in[20];
    const float* ic0_b    = (const float*)d_in[21];
    const float* ic1_k    = (const float*)d_in[22];
    const float* ic1_b    = (const float*)d_in[23];

    float* out = (float*)d_out;                 // (B, VOCAB, T-1)

    // ---- workspace layout ----
    float* ws = (float*)d_ws;
    float* mean_buf = ws;                                   // B*F
    float* tmp_buf  = mean_buf + (size_t)BB * FF;           // B*U
    float* h_buf    = tmp_buf  + (size_t)BB * UU;           // B*H
    float* c_buf    = h_buf    + (size_t)BB * HH;           // B*H
    float* fp_buf   = c_buf    + (size_t)BB * HH;           // (B*L)*U
    float* hp_buf   = fp_buf   + (size_t)BB * LL * UU;      // B*U
    float* z_buf    = hp_buf   + (size_t)BB * UU;           // B*4H
    float* f32_end  = z_buf    + (size_t)BB * 4 * HH;

    __bf16* bws      = (__bf16*)f32_end;                    // 16B-aligned
    __bf16* xcat_bf  = bws;                                 // B*XW
    __bf16* hbf      = xcat_bf + (size_t)BB * XW;           // B*H
    __bf16* W2t      = hbf     + (size_t)BB * HH;           // U*H   (ldt=H)
    __bf16* WcatT    = W2t     + (size_t)UU * HH;           // 4H*XW (ldt=XW)
    __bf16* LT       = WcatT   + (size_t)4 * HH * XW;       // VOCAB*XW
    __bf16* scratchT = LT      + (size_t)VOCAB * XW;        // max 512*2048

    // ---- one-time precompute ----
    mean_pool_kernel<<<(BB * FF + 255) / 256, 256, 0, stream>>>(img, mean_buf);

    // init h = tanh(relu(relu(mean@ih0+b)@ih1+b)), init c likewise
    tconv(stream, ih0_k, scratchT, FF, HH, FF, 0);
    gemm(stream, mean_buf, FF, 0, scratchT, FF, ih0_b, tmp_buf,
         BB, HH, FF, /*relu*/1, HH, 1, 0);
    tconv(stream, ih1_k, scratchT, HH, HH, HH, 0);
    gemm(stream, tmp_buf, HH, 0, scratchT, HH, ih1_b, h_buf,
         BB, HH, HH, /*tanh(relu)*/3, HH, 1, 0);
    tconv(stream, ic0_k, scratchT, FF, HH, FF, 0);
    gemm(stream, mean_buf, FF, 0, scratchT, FF, ic0_b, tmp_buf,
         BB, HH, FF, 1, HH, 1, 0);
    tconv(stream, ic1_k, scratchT, HH, HH, HH, 0);
    gemm(stream, tmp_buf, HH, 0, scratchT, HH, ic1_b, c_buf,
         BB, HH, HH, 3, HH, 1, 0);

    // feat_proj = img(B*L, F) @ W1 + b -> (B*L, U)
    tconv(stream, W1_k, scratchT, FF, UU, FF, 0);
    gemm(stream, img, FF, 0, scratchT, FF, W1_b, fp_buf,
         BB * LL, UU, FF, 0, UU, 1, 0);

    seed_h_kernel<<<(BB * HH + 255) / 256, 256, 0, stream>>>(h_buf, hbf, xcat_bf);

    // persistent transposed bf16 weights (reused by all 19 steps; L2-resident)
    tconv(stream, W2_k, W2t, HH, UU, HH, 0);
    tconv(stream, lstm_k,  WcatT, EDIM + FF, 4 * HH, XW, 0);       // rows 0..2303
    tconv(stream, lstm_rk, WcatT, HH,        4 * HH, XW, EDIM + FF); // rows 2304..
    tconv(stream, logit_k, LT, XW, VOCAB, XW, 0);

    // ---- decode loop (teacher forcing) ----
    for (int t = 0; t < NSTEP; ++t) {
        embed_kernel<<<BB, EDIM, 0, stream>>>(target, emb, xcat_bf, t);

        // hid_proj = h @ W2 + b
        gemm(stream, hbf, HH, 1, W2t, HH, W2_b, hp_buf,
             BB, UU, HH, 0, UU, 1, 0);

        attention_kernel<<<BB, 256, 0, stream>>>(fp_buf, hp_buf, V_k, V_b,
                                                 fbeta_k, fbeta_b, h_buf, img,
                                                 xcat_bf);

        // z = [e,ctx,h] @ [lstm_k; lstm_rk] + lstm_b   (single fused GEMM)
        gemm(stream, xcat_bf, XW, 1, WcatT, XW, lstm_b, z_buf,
             BB, 4 * HH, XW, 0, 4 * HH, 1, 0);

        lstm_gate_kernel<<<(BB * HH + 255) / 256, 256, 0, stream>>>(
            z_buf, h_buf, c_buf, hbf, xcat_bf);

        // logits = [e,ctx,h2] @ logit_k + logit_b, stored at out[b, v, t]
        gemm(stream, xcat_bf, XW, 1, LT, XW, logit_b, out,
             BB, VOCAB, XW, 0, (long long)VOCAB * NSTEP, NSTEP, t);
    }
}
